// TransformerBlock_64639257805460
// MI455X (gfx1250) — matrix operand order repaired and checked by reading, not compiled, not run on hardware
//
#include <hip/hip_runtime.h>
#include <hip/hip_bf16.h>
#include <stdint.h>

#define H_DIM   1024
#define S_LEN   2048
#define BATCH   2
#define NHEAD   16
#define HEADD   64
#define FFN_DIM 4096
#define MROWS   (BATCH * S_LEN)   // 4096
#define EPSLN   1e-5f

typedef _Float16 v16h __attribute__((ext_vector_type(16)));
typedef float    v8f  __attribute__((ext_vector_type(8)));
typedef unsigned int u32x4 __attribute__((ext_vector_type(4)));
typedef int          i32x8 __attribute__((ext_vector_type(8)));
typedef int          i32x4 __attribute__((ext_vector_type(4)));

union Frag16 {
    v16h     v;
    uint32_t u[8];
    uint4    q[2];
};

// ---------------------------------------------------------------- utilities
__global__ void f32_to_f16_kernel(const float* __restrict__ in,
                                  _Float16* __restrict__ out, int n) {
    for (int i = blockIdx.x * blockDim.x + threadIdx.x; i < n;
         i += gridDim.x * blockDim.x) {
        out[i] = (_Float16)in[i];
    }
}

// LayerNorm over last dim (1024); one block (256 threads) per row.
__global__ void layernorm_f16_kernel(const float* __restrict__ x,
                                     const float* __restrict__ g,
                                     const float* __restrict__ b,
                                     _Float16* __restrict__ out) {
    __shared__ float s_sum[256];
    __shared__ float s_sq[256];
    const int row = blockIdx.x;
    const int tid = threadIdx.x;
    const float* xr = x + (size_t)row * H_DIM;

    float lsum = 0.f, lsq = 0.f;
    float vals[4];
#pragma unroll
    for (int k = 0; k < 4; ++k) {
        float v = xr[tid + 256 * k];
        vals[k] = v;
        lsum += v;
        lsq  += v * v;
    }
    s_sum[tid] = lsum;
    s_sq[tid]  = lsq;
    __syncthreads();
    for (int off = 128; off > 0; off >>= 1) {
        if (tid < off) {
            s_sum[tid] += s_sum[tid + off];
            s_sq[tid]  += s_sq[tid + off];
        }
        __syncthreads();
    }
    const float mu  = s_sum[0] * (1.0f / H_DIM);
    const float var = s_sq[0] * (1.0f / H_DIM) - mu * mu;
    const float inv = rsqrtf(var + EPSLN);
    _Float16* orow = out + (size_t)row * H_DIM;
#pragma unroll
    for (int k = 0; k < 4; ++k) {
        int c = tid + 256 * k;
        orow[c] = (_Float16)((vals[k] - mu) * inv * g[c] + b[c]);
    }
}

// ---------------------------------------------------------------- WMMA GEMM
// C[M,N] = A[M,K](f16) @ B[K,N](f16) + bias, fused epilogue.
// 256 threads (8 waves), 128x128 tile; wave grid 2(M)x4(N), 4x2 wmma tiles.
// A-tile staged global->LDS by the Tensor Data Mover (wave 0 issues a 2D D#
// with pad: 64B rows + 16B pad => 80B LDS row stride == As[128][40]).
// EPI: 0 = scatter to q/k/v f16 buffers [B,NH,S,HD]
//      1 = f32 out with residual add
//      2 = exact GELU -> f16 out
template <int EPI>
__global__ __launch_bounds__(256)
void gemm_wmma_kernel(const _Float16* __restrict__ A,
                      const _Float16* __restrict__ B,
                      const float* __restrict__ bias,
                      int M, int N, int K,
                      float* __restrict__ outF32,
                      const float* __restrict__ res,
                      _Float16* __restrict__ outF16,
                      _Float16* __restrict__ q16,
                      _Float16* __restrict__ k16,
                      _Float16* __restrict__ v16) {
    __shared__ _Float16 As[128][40];   // [m][k], K-tile = 32, +8 pad (TDM pad)
    __shared__ _Float16 Bs[128][40];   // transposed: [n][k]

    const int tid  = threadIdx.x;
    const int wave = tid >> 5;
    const int lane = tid & 31;
    const int wm   = wave >> 2;        // 0..1
    const int wn   = wave & 3;         // 0..3
    const int m0   = blockIdx.y * 128;
    const int n0   = blockIdx.x * 128;
    const int lm   = lane & 15;
    const int lh   = lane >> 4;

    v8f acc[4][2];
#pragma unroll
    for (int mi = 0; mi < 4; ++mi)
#pragma unroll
        for (int ni = 0; ni < 2; ++ni) acc[mi][ni] = (v8f)0.0f;

    const uint32_t* Bg = (const uint32_t*)B;
    const int Nd = N >> 1;   // B row length in dwords

    // ---- loop-invariant TDM descriptor pieces (2D tile, 4B units)
    // group1: data_size=4B(2), pad_enable, pad_interval=16dw(3), pad_amount=4dw(3)
    const uint32_t ldsA = (uint32_t)(uintptr_t)&As[0][0];
    i32x8 g1;
    g1[0] = (int)((2u << 16) | (1u << 20) | (3u << 22) | (3u << 25));
    g1[1] = (int)(16u << 16);           // tensor_dim0 = 16 dwords (low16 @bit48)
    g1[2] = (int)(128u << 16);          // tensor_dim1 = 128 rows
    g1[3] = (int)(16u << 16);           // tile_dim0 = 16 dwords
    g1[4] = 128;                        // tile_dim1 = 128, tile_dim2 = 0
    g1[5] = (int)(uint32_t)(K >> 1);    // tensor_dim0_stride (dwords)
    g1[6] = 0;
    g1[7] = 0;
    const i32x4 zero4 = {0, 0, 0, 0};
    const i32x8 zero8 = {0, 0, 0, 0, 0, 0, 0, 0};

    for (int k0 = 0; k0 < K; k0 += 32) {
        // ---- TDM: stage A tile 128x(32 halves) into As with padding
        if (wave == 0) {
            uint64_t ga = (uint64_t)(uintptr_t)A + ((size_t)m0 * K + k0) * 2;
            u32x4 g0;
            g0[0] = 1u;  // count=1 (valid), load, not restore, no gather
            g0[1] = __builtin_amdgcn_readfirstlane(ldsA);
            g0[2] = __builtin_amdgcn_readfirstlane((uint32_t)ga);
            g0[3] = __builtin_amdgcn_readfirstlane(
                        ((uint32_t)(ga >> 32) & 0x1FFFFFFu) | (2u << 30));
            __builtin_amdgcn_tensor_load_to_lds(g0, g1, zero4, zero4, zero8, 0);
        }
        // ---- stage B tile 32x128 transposed -> Bs[n][k] (manual: needs T)
#pragma unroll
        for (int i = 0; i < 8; ++i) {
            int idx = i * 256 + tid;
            int kr = idx >> 6, nc = idx & 63;
            union { uint32_t u; _Float16 h[2]; } cv;
            cv.u = Bg[(size_t)(k0 + kr) * Nd + (n0 >> 1) + nc];
            Bs[nc * 2][kr]     = cv.h[0];
            Bs[nc * 2 + 1][kr] = cv.h[1];
        }
        // ---- prefetch next B tile into caches
        if (k0 + 32 < K) {
            __builtin_prefetch(
                &Bg[(size_t)(k0 + 32) * Nd + (n0 >> 1) + (tid & 63)], 0, 3);
        }
        if (wave == 0) {
            __builtin_amdgcn_s_wait_tensorcnt(0);
        }
        __syncthreads();

        // ---- fragments
        Frag16 af[4];
#pragma unroll
        for (int mi = 0; mi < 4; ++mi) {
            int m = wm * 64 + mi * 16 + lm;
#pragma unroll
            for (int d = 0; d < 8; ++d) {
                int kk = (d >> 2) * 16 + lh * 8 + (d & 3) * 2;
                af[mi].u[d] = *(const uint32_t*)&As[m][kk];
            }
        }
        Frag16 bf[2];
#pragma unroll
        for (int ni = 0; ni < 2; ++ni) {
            int n = wn * 32 + ni * 16 + lm;
            bf[ni].q[0] = *(const uint4*)&Bs[n][lh * 8];
            bf[ni].q[1] = *(const uint4*)&Bs[n][16 + lh * 8];
        }
#pragma unroll
        for (int mi = 0; mi < 4; ++mi)
#pragma unroll
            for (int ni = 0; ni < 2; ++ni)
                acc[mi][ni] = __builtin_amdgcn_wmma_f32_16x16x32_f16(
                    false, af[mi].v, false, bf[ni].v,
                    (short)0, acc[mi][ni], false, false);
        __syncthreads();
    }

    // ---- epilogue
#pragma unroll
    for (int mi = 0; mi < 4; ++mi) {
#pragma unroll
        for (int ni = 0; ni < 2; ++ni) {
#pragma unroll
            for (int i = 0; i < 8; ++i) {
                int gm = m0 + wm * 64 + mi * 16 + lh * 8 + i;
                int gn = n0 + wn * 32 + ni * 16 + lm;
                float v = acc[mi][ni][i] + bias[gn];
                if (EPI == 0) {
                    int c3   = gn >> 10;        // 0:q 1:k 2:v
                    int rem  = gn & 1023;
                    int head = rem >> 6;
                    int hd   = rem & 63;
                    int bb   = gm >> 11;        // /S_LEN
                    int ss   = gm & 2047;
                    _Float16* dst = (c3 == 0) ? q16 : (c3 == 1) ? k16 : v16;
                    dst[((((size_t)bb * NHEAD + head) * S_LEN + ss) * HEADD) + hd] =
                        (_Float16)v;
                } else if (EPI == 1) {
                    size_t o = (size_t)gm * N + gn;
                    outF32[o] = v + res[o];
                } else {  // EPI == 2, exact GELU
                    float g = 0.5f * v * (1.0f + erff(v * 0.70710678118654752f));
                    outF16[(size_t)gm * N + gn] = (_Float16)g;
                }
            }
        }
    }
}

// ---------------------------------------------------------- Flash attention
// One wave per 16-query tile; workgroup = 8 waves = 128 queries of one (b,h).
// Per-wave private LDS regions (no barriers; in-wave DS ordering suffices).
__global__ __launch_bounds__(256)
void flash_attn_kernel(const _Float16* __restrict__ q16,
                       const _Float16* __restrict__ k16,
                       const _Float16* __restrict__ v16,
                       _Float16* __restrict__ attn16) {
    __shared__ _Float16 Vt[8][64 * 40];   // per wave: [hd][kv], kv-chunk=32, pad 8
    __shared__ _Float16 Pt[8][16 * 40];   // per wave: [m][k]

    const int tid  = threadIdx.x;
    const int wave = tid >> 5;
    const int lane = tid & 31;
    const int lm   = lane & 15;
    const int lh   = lane >> 4;
    const int h    = blockIdx.y;
    const int b    = blockIdx.z;
    const int qtile = blockIdx.x * 128 + wave * 16;
    const size_t bh = (size_t)b * NHEAD + h;

    _Float16* vts = Vt[wave];
    _Float16* pts = Pt[wave];

    // ---- Q fragments (features 0..31 and 32..63)
    Frag16 qa[2];
    {
        const _Float16* qr = q16 + (bh * S_LEN + qtile + lm) * HEADD;
#pragma unroll
        for (int fb = 0; fb < 2; ++fb) {
            qa[fb].q[0] = *(const uint4*)(qr + fb * 32 + lh * 8);
            qa[fb].q[1] = *(const uint4*)(qr + fb * 32 + 16 + lh * 8);
        }
    }

    v8f o[4];
#pragma unroll
    for (int t = 0; t < 4; ++t) o[t] = (v8f)0.0f;
    float mrow[8], lrow[8];
#pragma unroll
    for (int i = 0; i < 8; ++i) { mrow[i] = -INFINITY; lrow[i] = 0.0f; }

    const int nchunk = (qtile + 15) / 32 + 1;   // kv chunks of 32 (causal bound)
    const float scale = 0.125f;                  // 1/sqrt(64)

    for (int j = 0; j < nchunk; ++j) {
        const int kv0 = 32 * j;

        // ---- scores: two 16x16 tiles, each = Q(16x64) @ K^T via 2 wmma
        v8f s[2];
#pragma unroll
        for (int t = 0; t < 2; ++t) {
            const _Float16* kr =
                k16 + (bh * S_LEN + kv0 + t * 16 + lm) * HEADD;
            Frag16 kb0, kb1;
            kb0.q[0] = *(const uint4*)(kr + lh * 8);
            kb0.q[1] = *(const uint4*)(kr + 16 + lh * 8);
            kb1.q[0] = *(const uint4*)(kr + 32 + lh * 8);
            kb1.q[1] = *(const uint4*)(kr + 32 + 16 + lh * 8);
            v8f sv = (v8f)0.0f;
            sv = __builtin_amdgcn_wmma_f32_16x16x32_f16(false, qa[0].v, false,
                                                        kb0.v, (short)0, sv,
                                                        false, false);
            sv = __builtin_amdgcn_wmma_f32_16x16x32_f16(false, qa[1].v, false,
                                                        kb1.v, (short)0, sv,
                                                        false, false);
            s[t] = sv;
        }

        // ---- scale + causal mask
        float s0[8], s1[8];
#pragma unroll
        for (int i = 0; i < 8; ++i) {
            int row = qtile + lh * 8 + i;
            int c0 = kv0 + lm;
            int c1 = kv0 + 16 + lm;
            s0[i] = (c0 > row) ? -INFINITY : s[0][i] * scale;
            s1[i] = (c1 > row) ? -INFINITY : s[1][i] * scale;
        }

        // ---- online softmax row stats (reduce across the 16-lane group)
        float alpha[8], psum[8];
#pragma unroll
        for (int i = 0; i < 8; ++i) {
            float cand = fmaxf(s0[i], s1[i]);
            for (int off = 1; off < 16; off <<= 1)
                cand = fmaxf(cand, __shfl_xor(cand, off));
            float mnew = fmaxf(mrow[i], cand);
            alpha[i] = expf(mrow[i] - mnew);
            float p0 = expf(s0[i] - mnew);
            float p1 = expf(s1[i] - mnew);
            s0[i] = p0;
            s1[i] = p1;
            float su = p0 + p1;
            for (int off = 1; off < 16; off <<= 1)
                su += __shfl_xor(su, off);
            psum[i] = su;
            mrow[i] = mnew;
        }
#pragma unroll
        for (int i = 0; i < 8; ++i) lrow[i] = lrow[i] * alpha[i] + psum[i];

        // ---- stage P (16x32, f16) into per-wave LDS
#pragma unroll
        for (int i = 0; i < 8; ++i) {
            int row = lh * 8 + i;
            pts[row * 40 + lm]      = (_Float16)s0[i];
            pts[row * 40 + 16 + lm] = (_Float16)s1[i];
        }
        // ---- stage V chunk 32x64 transposed: vts[hd][kv]
        {
            const uint32_t* vr =
                (const uint32_t*)(v16 + (bh * S_LEN + kv0 + lane) * HEADD);
#pragma unroll
            for (int c = 0; c < 32; ++c) {
                union { uint32_t u; _Float16 hh[2]; } cv;
                cv.u = vr[c];
                vts[(2 * c) * 40 + lane]     = cv.hh[0];
                vts[(2 * c + 1) * 40 + lane] = cv.hh[1];
            }
        }

        // ---- rescale O, then O += P @ V
#pragma unroll
        for (int t = 0; t < 4; ++t)
#pragma unroll
            for (int i = 0; i < 8; ++i) o[t][i] *= alpha[i];

        Frag16 pa;
#pragma unroll
        for (int d = 0; d < 8; ++d) {
            int kk = (d >> 2) * 16 + lh * 8 + (d & 3) * 2;
            pa.u[d] = *(const uint32_t*)&pts[lm * 40 + kk];
        }
#pragma unroll
        for (int t = 0; t < 4; ++t) {
            int hd = t * 16 + lm;
            Frag16 vb;
            vb.q[0] = *(const uint4*)&vts[hd * 40 + lh * 8];
            vb.q[1] = *(const uint4*)&vts[hd * 40 + 16 + lh * 8];
            o[t] = __builtin_amdgcn_wmma_f32_16x16x32_f16(
                false, pa.v, false, vb.v, (short)0, o[t], false, false);
        }
    }

    // ---- normalize and store (attn16 layout [B,S,H], heads concatenated)
#pragma unroll
    for (int i = 0; i < 8; ++i) {
        float inv = 1.0f / lrow[i];
        int grow = qtile + lh * 8 + i;
        _Float16* orow =
            attn16 + ((size_t)b * S_LEN + grow) * H_DIM + h * HEADD;
#pragma unroll
        for (int t = 0; t < 4; ++t)
            orow[t * 16 + lm] = (_Float16)(o[t][i] * inv);
    }
}

// ---------------------------------------------------------------- launcher
extern "C" void kernel_launch(void* const* d_in, const int* in_sizes, int n_in,
                              void* d_out, int out_size, void* d_ws,
                              size_t ws_size, hipStream_t stream) {
    const float* x     = (const float*)d_in[0];
    const float* w_qkv = (const float*)d_in[1];
    const float* b_qkv = (const float*)d_in[2];
    const float* w_out = (const float*)d_in[3];
    const float* b_out = (const float*)d_in[4];
    const float* ln1_g = (const float*)d_in[5];
    const float* ln1_b = (const float*)d_in[6];
    const float* ln2_g = (const float*)d_in[7];
    const float* ln2_b = (const float*)d_in[8];
    const float* w1    = (const float*)d_in[9];
    const float* b1    = (const float*)d_in[10];
    const float* w2    = (const float*)d_in[11];
    const float* b2    = (const float*)d_in[12];
    float* out = (float*)d_out;

    char* base = (char*)d_ws;
    size_t off = 0;
    auto alloc = [&](size_t bytes) {
        size_t o = off;
        off = (off + bytes + 255) & ~(size_t)255;
        return o;
    };
    const size_t nWqkv = (size_t)H_DIM * 3 * H_DIM;
    const size_t nWout = (size_t)H_DIM * H_DIM;
    const size_t nW1   = (size_t)H_DIM * FFN_DIM;
    const size_t nW2   = (size_t)FFN_DIM * H_DIM;
    const size_t nAct  = (size_t)MROWS * H_DIM;
    const size_t nHead = (size_t)BATCH * NHEAD * S_LEN * HEADD;

    _Float16* wqkv16 = (_Float16*)(base + alloc(nWqkv * 2));
    _Float16* wout16 = (_Float16*)(base + alloc(nWout * 2));
    _Float16* w1_16  = (_Float16*)(base + alloc(nW1 * 2));
    _Float16* w2_16  = (_Float16*)(base + alloc(nW2 * 2));
    _Float16* h1_16  = (_Float16*)(base + alloc(nAct * 2));
    _Float16* q16    = (_Float16*)(base + alloc(nHead * 2));
    _Float16* k16    = (_Float16*)(base + alloc(nHead * 2));
    _Float16* v16    = (_Float16*)(base + alloc(nHead * 2));
    _Float16* attn16 = (_Float16*)(base + alloc(nAct * 2));
    float*    x1     = (float*)(base + alloc(nAct * 4));
    _Float16* h2_16  = (_Float16*)(base + alloc(nAct * 2));
    _Float16* act16  = (_Float16*)(base + alloc((size_t)MROWS * FFN_DIM * 2));

    // 1) weight conversions fp32 -> f16
    f32_to_f16_kernel<<<2048, 256, 0, stream>>>(w_qkv, wqkv16, (int)nWqkv);
    f32_to_f16_kernel<<<1024, 256, 0, stream>>>(w_out, wout16, (int)nWout);
    f32_to_f16_kernel<<<2048, 256, 0, stream>>>(w1, w1_16, (int)nW1);
    f32_to_f16_kernel<<<2048, 256, 0, stream>>>(w2, w2_16, (int)nW2);

    // 2) LN1
    layernorm_f16_kernel<<<MROWS, 256, 0, stream>>>(x, ln1_g, ln1_b, h1_16);

    // 3) QKV projection: (4096x1024)@(1024x3072) + bias, scatter q/k/v
    gemm_wmma_kernel<0><<<dim3(3 * H_DIM / 128, MROWS / 128), 256, 0, stream>>>(
        h1_16, wqkv16, b_qkv, MROWS, 3 * H_DIM, H_DIM,
        nullptr, nullptr, nullptr, q16, k16, v16);

    // 4) causal flash attention
    flash_attn_kernel<<<dim3(S_LEN / 128, NHEAD, BATCH), 256, 0, stream>>>(
        q16, k16, v16, attn16);

    // 5) output projection + residual: x1 = attn@w_out + b_out + x
    gemm_wmma_kernel<1><<<dim3(H_DIM / 128, MROWS / 128), 256, 0, stream>>>(
        attn16, wout16, b_out, MROWS, H_DIM, H_DIM,
        x1, x, nullptr, nullptr, nullptr, nullptr);

    // 6) LN2
    layernorm_f16_kernel<<<MROWS, 256, 0, stream>>>(x1, ln2_g, ln2_b, h2_16);

    // 7) FFN up + exact GELU: act = gelu(h2@w1 + b1)
    gemm_wmma_kernel<2><<<dim3(FFN_DIM / 128, MROWS / 128), 256, 0, stream>>>(
        h2_16, w1_16, b1, MROWS, FFN_DIM, H_DIM,
        nullptr, nullptr, act16, nullptr, nullptr, nullptr);

    // 8) FFN down + residual: out = act@w2 + b2 + x1
    gemm_wmma_kernel<1><<<dim3(H_DIM / 128, MROWS / 128), 256, 0, stream>>>(
        act16, w2_16, b2, MROWS, H_DIM, FFN_DIM,
        out, x1, nullptr, nullptr, nullptr, nullptr);
}
